// FFMDouble_5772436045842
// MI455X (gfx1250) — matrix operand order repaired and checked by reading, not compiled, not run on hardware
//
#include <hip/hip_runtime.h>
#include <hip/hip_bf16.h>

// ---------------------------------------------------------------------------
// FFM double block for MI455X (gfx1250, wave32, WMMA, async LDS copies).
//   B=32, T=1024, IN=HID=OUT=512, MEM=128, C2=8, PRE_OUT=1280, MIX_IN=2048
// All GEMM operands staged as f16 (halves repeated HBM reads); WMMA f32 accum.
// Inner GEMM loop: async global->LDS (ASYNCcnt) + ds_load_b128 + v_wmma only.
// ---------------------------------------------------------------------------

#define BT_TOTAL 32768      // B*T
#define DIM      512
#define MEMN     128
#define C2N      8
#define PRE_OUT  1280
#define MIX_IN   2048

typedef __attribute__((ext_vector_type(16))) _Float16 v16h;
typedef __attribute__((ext_vector_type(8)))  _Float16 v8h;
typedef __attribute__((ext_vector_type(2)))  _Float16 v2h;
typedef __attribute__((ext_vector_type(8)))  float    v8f;

#define BM 128
#define BN 128
#define BK 32

// ---- CDNA5 async global->LDS copy (ASYNCcnt-tracked), 16B per lane --------
__device__ __forceinline__ void cp_async_h8(_Float16* lds_dst, const _Float16* gsrc)
{
#if defined(__gfx1250__)
    unsigned ldsOff = (unsigned)(size_t)lds_dst;            // LDS aperture low32
    unsigned long long ga = (unsigned long long)(size_t)gsrc;
    asm volatile("global_load_async_to_lds_b128 %0, %1, off"
                 :: "v"(ldsOff), "v"(ga) : "memory");
#else
    *(float4*)(void*)lds_dst = *(const float4*)(const void*)gsrc;  // host pass
#endif
}

__device__ __forceinline__ void async_wait_all()
{
#if defined(__gfx1250__)
    asm volatile("s_wait_asynccnt 0x0" ::: "memory");
#endif
}

// f32 -> f16 bulk convert, 8 elements/thread (n must be a multiple of 8)
__global__ __launch_bounds__(256) void to_f16(
    const float* __restrict__ s, _Float16* __restrict__ d, size_t n8)
{
    size_t i = (size_t)blockIdx.x * 256 + threadIdx.x;
    if (i >= n8) return;
    const float4* sp = (const float4*)(s + i * 8);
    float4 v0 = sp[0], v1 = sp[1];
    v8h o;
    o[0] = (_Float16)v0.x; o[1] = (_Float16)v0.y;
    o[2] = (_Float16)v0.z; o[3] = (_Float16)v0.w;
    o[4] = (_Float16)v1.x; o[5] = (_Float16)v1.y;
    o[6] = (_Float16)v1.z; o[7] = (_Float16)v1.w;
    *(v8h*)(d + i * 8) = o;
}

// C[M,N] = A[M,K] * W[N,K]^T + bias[N]   (A,W f16; C f32)
__global__ __launch_bounds__(256) void gemm_f16wmma(
    const _Float16* __restrict__ A, const _Float16* __restrict__ W,
    const float* __restrict__ bias, float* __restrict__ C,
    int M, int N, int K)
{
    // double-buffered f16 tiles: 2 * (8KB + 8KB) = 32KB LDS
    __shared__ __align__(16) _Float16 Ash[2][BM][BK];
    __shared__ __align__(16) _Float16 Wsh[2][BN][BK];

    const int tid  = threadIdx.x;
    const int lane = tid & 31;
    const int wave = tid >> 5;
    const int mBlk = blockIdx.x * BM;
    const int nBlk = blockIdx.y * BN;

    // 8 waves -> 4 (m) x 2 (n); each wave: 2 m-tiles x 4 n-tiles of 16x16
    const int wm = wave >> 1;   // 0..3
    const int wn = wave & 1;    // 0..1

    v8f acc[2][4] = {};

    // loader: 2 threads per row; each thread 2x 16B (8-f16) async copies/tile
    const int ldRow = tid >> 1;
    const int ldCol = (tid & 1) * 16;   // f16 elements
    const _Float16* aRow = A + (size_t)(mBlk + ldRow) * K + ldCol;
    const _Float16* wRow = W + (size_t)(nBlk + ldRow) * K + ldCol;

    auto issueTiles = [&](int bsel, int kk) {
#pragma unroll
        for (int q = 0; q < 2; ++q) {
            cp_async_h8(&Ash[bsel][ldRow][ldCol + 8 * q], aRow + kk + 8 * q);
            cp_async_h8(&Wsh[bsel][ldRow][ldCol + 8 * q], wRow + kk + 8 * q);
        }
    };

    issueTiles(0, 0);            // prologue: tile 0 in flight
    int buf = 0;

    for (int k0 = 0; k0 < K; k0 += BK) {
        async_wait_all();        // my async copies landed in LDS
        __syncthreads();         // everyone's copies visible (+ prior reads done)

        if (k0 + BK < K)
            issueTiles(buf ^ 1, k0 + BK);       // stream next tile during compute
        if (k0 + 2 * BK < K) {                  // warm L2 one more tile ahead
            __builtin_prefetch(aRow + k0 + 2 * BK, 0, 0);
            __builtin_prefetch(wRow + k0 + 2 * BK, 0, 0);
        }

#if defined(__gfx1250__)
        const int mrow  = lane & 15;
        const int half  = lane >> 4;
        const int khalf = half * 16;

        // A fragments: 16x32 ISA layout == two aligned b128 runs per lane
        v16h afrag[2];
#pragma unroll
        for (int i = 0; i < 2; ++i) {
            const _Float16* ar = &Ash[buf][wm * 32 + i * 16 + mrow][0];
            v8h lo = *(const v8h*)(ar + half * 8);
            v8h hi = *(const v8h*)(ar + 16 + half * 8);
            afrag[i] = __builtin_shufflevector(lo, hi,
                0, 1, 2, 3, 4, 5, 6, 7, 8, 9, 10, 11, 12, 13, 14, 15);
        }
        // B fragments: 32x16; lane n reads W row n in the same K order as A
        v16h bfrag[4];
#pragma unroll
        for (int j = 0; j < 4; ++j) {
            const _Float16* wr = &Wsh[buf][wn * 64 + j * 16 + mrow][0];
            v8h lo = *(const v8h*)(wr + half * 8);
            v8h hi = *(const v8h*)(wr + 16 + half * 8);
            bfrag[j] = __builtin_shufflevector(lo, hi,
                0, 1, 2, 3, 4, 5, 6, 7, 8, 9, 10, 11, 12, 13, 14, 15);
        }
#pragma unroll
        for (int i = 0; i < 2; ++i)
#pragma unroll
            for (int j = 0; j < 4; ++j)
                acc[i][j] = __builtin_amdgcn_wmma_f32_16x16x32_f16(
                    false, afrag[i], false, bfrag[j],
                    (short)0, acc[i][j], false, false);
#endif
        buf ^= 1;
    }

    // epilogue: C/D 16x16 layout (lanes 0-15 -> M=r, lanes 16-31 -> M=8+r)
    const int nlane = lane & 15;
    const int mhalf = (lane >> 4) * 8;
#pragma unroll
    for (int j = 0; j < 4; ++j) {
        const int n  = nBlk + wn * 64 + j * 16 + nlane;
        const float bv = bias[n];
#pragma unroll
        for (int i = 0; i < 2; ++i) {
            const int mBase = mBlk + wm * 32 + i * 16 + mhalf;
#pragma unroll
            for (int r = 0; r < 8; ++r)
                C[(size_t)(mBase + r) * N + n] = acc[i][j][r] + bv;
        }
    }
}

// u[bt,m] = p[bt,m] * sigmoid(p[bt, 640+m])
__global__ __launch_bounds__(256) void gate_mul(
    const float* __restrict__ p, float* __restrict__ u)
{
    size_t i  = (size_t)blockIdx.x * blockDim.x + threadIdx.x;  // BT*128
    size_t bt = i >> 7;
    int    m  = (int)(i & 127);
    const float* row = p + bt * PRE_OUT;
    float y = row[m];
    float g = row[MEMN + DIM + m];              // 640 + m
    u[i] = y / (1.0f + expf(-g));
}

// one thread per (b, m, c): s = gamma*s + u.
// Writes f32 complex states to d_out AND an f16 copy feeding the mix GEMM.
__global__ __launch_bounds__(256) void ffa_scan(
    const float* __restrict__ u, const float* __restrict__ av,
    const float* __restrict__ bv, float* __restrict__ st,
    _Float16* __restrict__ st16)
{
    int tid = blockIdx.x * 256 + threadIdx.x;   // 0..32767
    int b = tid >> 10;
    int m = (tid >> 3) & 127;
    int c = tid & 7;

    float decay = expf(-fabsf(av[m]));
    float ph    = bv[c];
    float gr = decay * cosf(ph);
    float gi = decay * sinf(ph);

    float sr = 0.0f, si = 0.0f;
    const float* up = u + (size_t)b * 1024 * MEMN + m;
    size_t base = ((size_t)b * 1024 * MEMN + m) * (C2N * 2) + c * 2;
    float*     sp  = st + base;
    _Float16*  hp  = st16 + base;
    for (int t = 0; t < 1024; ++t) {
        float y  = up[(size_t)t * MEMN];
        float nr = fmaf(gr, sr, fmaf(-gi, si, y));
        float ni = fmaf(gr, si, gi * sr);
        sr = nr; si = ni;
        __builtin_nontemporal_store(sr, sp);
        __builtin_nontemporal_store(si, sp + 1);
        v2h hpair;
        hpair[0] = (_Float16)sr;
        hpair[1] = (_Float16)si;
        __builtin_nontemporal_store(hpair, (v2h*)hp);
        sp += (size_t)MEMN * C2N * 2;
        hp += (size_t)MEMN * C2N * 2;
    }
}

// out = LN(z)*sigmoid(p[768+n]) + p[128+n]*(1-sigmoid); 1 wave per row.
// writeF32 != 0: write f32 to outf (final output); else write f16 to out16
// (intermediate h for the next block's pre-GEMM).
__global__ __launch_bounds__(256) void mix_post(
    const float* __restrict__ z, const float* __restrict__ p,
    _Float16* __restrict__ out16, float* __restrict__ outf, int writeF32)
{
    int wave = threadIdx.x >> 5;
    int lane = threadIdx.x & 31;
    size_t row = (size_t)blockIdx.x * 8 + wave;

    const float* zr = z + row * DIM;
    float vals[16];
    float s = 0.0f, ss = 0.0f;
#pragma unroll
    for (int j = 0; j < 16; ++j) {
        float v = zr[lane + 32 * j];
        vals[j] = v;
        s  += v;
        ss += v * v;
    }
#pragma unroll
    for (int off = 16; off > 0; off >>= 1) {
        s  += __shfl_xor(s, off, 32);
        ss += __shfl_xor(ss, off, 32);
    }
    float mean = s * (1.0f / (float)DIM);
    float var  = ss * (1.0f / (float)DIM) - mean * mean;
    float rstd = rsqrtf(var + 1e-5f);

    const float* pr = p + row * PRE_OUT;
#pragma unroll
    for (int j = 0; j < 16; ++j) {
        int n = lane + 32 * j;
        float og   = 1.0f / (1.0f + expf(-pr[2 * MEMN + DIM + n]));  // 768+n
        float thru = pr[MEMN + n];                                   // 128+n
        float val  = (vals[j] - mean) * rstd * og + thru * (1.0f - og);
        if (writeF32) outf[row * DIM + n] = val;
        else          out16[row * DIM + n] = (_Float16)val;
    }
}

extern "C" void kernel_launch(void* const* d_in, const int* in_sizes, int n_in,
                              void* d_out, int out_size, void* d_ws, size_t ws_size,
                              hipStream_t stream)
{
    (void)in_sizes; (void)n_in; (void)out_size; (void)ws_size;

    const float* x      = (const float*)d_in[0];
    const float* pre0_w = (const float*)d_in[1];
    const float* pre0_b = (const float*)d_in[2];
    const float* pre1_w = (const float*)d_in[3];
    const float* pre1_b = (const float*)d_in[4];
    const float* mix0_w = (const float*)d_in[5];
    const float* mix0_b = (const float*)d_in[6];
    const float* mix1_w = (const float*)d_in[7];
    const float* mix1_b = (const float*)d_in[8];
    const float* a0 = (const float*)d_in[9];
    const float* b0 = (const float*)d_in[10];
    const float* a1 = (const float*)d_in[11];
    const float* b1 = (const float*)d_in[12];

    // d_out: out1 | state0 (interleaved complex) | state1
    float* out1 = (float*)d_out;
    float* st0  = out1 + (size_t)BT_TOTAL * DIM;
    float* st1  = st0  + (size_t)BT_TOTAL * MIX_IN;

    // workspace (float-sized slots): p | u | z | h16 | st16 | 4x weight f16
    float* p  = (float*)d_ws;
    float* u  = p  + (size_t)BT_TOTAL * PRE_OUT;
    float* z  = u  + (size_t)BT_TOTAL * MEMN;
    _Float16* h16  = (_Float16*)(z + (size_t)BT_TOTAL * DIM);
    _Float16* s16  = h16 + (size_t)BT_TOTAL * DIM;       // BT x 2048 f16
    _Float16* pw0h = s16 + (size_t)BT_TOTAL * MIX_IN;
    _Float16* pw1h = pw0h + (size_t)PRE_OUT * DIM;
    _Float16* mw0h = pw1h + (size_t)PRE_OUT * DIM;
    _Float16* mw1h = mw0h + (size_t)DIM * MIX_IN;

    dim3 blk(256);
    dim3 gPre(BT_TOTAL / BM, PRE_OUT / BN);   // (256, 10)
    dim3 gMix(BT_TOTAL / BM, DIM / BN);       // (256, 4)
    dim3 gGate((BT_TOTAL * MEMN) / 256);      // 16384
    dim3 gScan(BT_TOTAL * C2N * MEMN / (C2N * MEMN * 256) * 1 + 127); // 255 (unused)
    dim3 gPost(BT_TOTAL / 8);                 // 4096

    const size_t nX8  = (size_t)BT_TOTAL * DIM / 8;      // 2,097,152
    const size_t nPW8 = (size_t)PRE_OUT * DIM / 8;       // 81,920
    const size_t nMW8 = (size_t)DIM * MIX_IN / 8;        // 131,072

    // one-time f16 staging of activations-in and weights
    to_f16<<<dim3((unsigned)((nX8  + 255) / 256)), blk, 0, stream>>>(x,      h16,  nX8);
    to_f16<<<dim3((unsigned)((nPW8 + 255) / 256)), blk, 0, stream>>>(pre0_w, pw0h, nPW8);
    to_f16<<<dim3((unsigned)((nPW8 + 255) / 256)), blk, 0, stream>>>(pre1_w, pw1h, nPW8);
    to_f16<<<dim3((unsigned)((nMW8 + 255) / 256)), blk, 0, stream>>>(mix0_w, mw0h, nMW8);
    to_f16<<<dim3((unsigned)((nMW8 + 255) / 256)), blk, 0, stream>>>(mix1_w, mw1h, nMW8);

    // ---- block 0 ----
    gemm_f16wmma<<<gPre, blk, 0, stream>>>(h16, pw0h, pre0_b, p, BT_TOTAL, PRE_OUT, DIM);
    gate_mul    <<<gGate, blk, 0, stream>>>(p, u);
    ffa_scan    <<<dim3(128), blk, 0, stream>>>(u, a0, b0, st0, s16);
    gemm_f16wmma<<<gMix, blk, 0, stream>>>(s16, mw0h, mix0_b, z, BT_TOTAL, DIM, MIX_IN);
    mix_post    <<<gPost, blk, 0, stream>>>(z, p, h16, (float*)nullptr, 0); // out0 -> f16

    // ---- block 1 ----
    gemm_f16wmma<<<gPre, blk, 0, stream>>>(h16, pw1h, pre1_b, p, BT_TOTAL, PRE_OUT, DIM);
    gate_mul    <<<gGate, blk, 0, stream>>>(p, u);
    ffa_scan    <<<dim3(128), blk, 0, stream>>>(u, a1, b1, st1, s16);
    gemm_f16wmma<<<gMix, blk, 0, stream>>>(s16, mw1h, mix1_b, z, BT_TOTAL, DIM, MIX_IN);
    mix_post    <<<gPost, blk, 0, stream>>>(z, p, (_Float16*)nullptr, out1, 1);
}
